// MambaSpotCore_82918638617162
// MI455X (gfx1250) — matrix operand order repaired and checked by reading, not compiled, not run on hardware
//
#include <hip/hip_runtime.h>
#include <hip/hip_bf16.h>
#include <stdint.h>

// ---------------- model constants ----------------
#define DEPTH    12
#define D_MODEL  192
#define D_INNER  384
#define N_STATE  16
#define DT_RANK  12
#define XP_N     48      // 44 padded to 48 (multiple of 16)
#define T_FR     8
#define NP_      196
#define L_SEQ    1576    // 8 * 197
#define BATCH    8
#define M_TOK    (BATCH * L_SEQ)       // 12608
#define M_PATCH  (BATCH * T_FR * NP_)  // 12544
#define K_PATCH  768                   // 3*16*16
#define NC_      18

typedef __attribute__((ext_vector_type(8)))  _Float16 v8h;
typedef __attribute__((ext_vector_type(16))) _Float16 v16h;
typedef __attribute__((ext_vector_type(8)))  float    v8f;

__device__ __forceinline__ float clip1e6(float v) {
    return fminf(fmaxf(v, -1000000.0f), 1000000.0f);
}
__device__ __forceinline__ float wave_red(float v) {
    #pragma unroll
    for (int o = 16; o > 0; o >>= 1) v += __shfl_xor(v, o, 32);
    return v;
}
__device__ __forceinline__ v16h cat16(v8h lo, v8h hi) {
    return __builtin_shufflevector(lo, hi, 0,1,2,3,4,5,6,7,8,9,10,11,12,13,14,15);
}

// ---------------- WMMA GEMM: async-LDS weight staging + K software pipeline ---
// C[M,N] = A[M,K] @ W[N,K]^T.  Block = 256 threads = 8 waves.
//   blockIdx.y selects an N-panel of TPW*16 columns; the (TPW*16 x KC) weight
//   panel (contiguous in W) is staged into LDS once via
//   GLOBAL_LOAD_ASYNC_TO_LDS_B128 and shared by all 8 waves.
//   Each wave computes 16(M) x TPW*16(N).  The K loop is double-buffered:
//   fragment loads for step k+1 are issued before the step-k WMMAs so the
//   load/ds counters drain behind the matrix ops instead of in front of them.
// POSTOP=1 fuses nan_to_num clamp on store.
#define GEMM_SMEM_HALFS 24576   // 48 KB static LDS, max over all instantiations

template <typename OUT_T, int POSTOP, int TPW, int KC>
__global__ void gemm_wmma_kernel(const _Float16* __restrict__ A,
                                 const _Float16* __restrict__ W,
                                 OUT_T* __restrict__ C,
                                 int M, int N) {
    __shared__ _Float16 smem[GEMM_SMEM_HALFS];
    constexpr int ROWS = TPW * 16;
    constexpr int NK   = KC >> 5;          // 32-wide K steps
    const int tid = threadIdx.x;

    // ---- stage the weight panel (ROWS x KC halfs, contiguous) into LDS ----
    {
        const _Float16* src = W + (size_t)blockIdx.y * ROWS * KC;
        constexpr int NOPS = (ROWS * KC) >> 3;            // 16-byte transfers
        const uint32_t lds0 = (uint32_t)(uintptr_t)(&smem[0]);
        #pragma unroll 1
        for (int op = tid; op < NOPS; op += 256) {
            uint32_t loff = lds0 + op * 16;
            uint64_t g = (uint64_t)(uintptr_t)(src + op * 8);
            asm volatile("global_load_async_to_lds_b128 %0, %1, off"
                         :: "v"(loff), "v"(g) : "memory");
        }
        asm volatile("s_wait_asynccnt 0x0" ::: "memory");
    }
    __syncthreads();

    const int wave = tid >> 5;
    const int lane = tid & 31;
    const int tm = blockIdx.x * 8 + wave;
    if (tm >= (M >> 4)) return;                           // uniform per wave
    const int r  = lane & 15;
    const int hs = lane >> 4;

    // A fragment: lane r holds row (tm*16+r); K halves {hs*8..+7, 16+hs*8..+7}
    const _Float16* a = A + ((size_t)(tm * 16 + r)) * KC + hs * 8;
    const _Float16* wbase = &smem[(size_t)r * KC + hs * 8];

    v8f acc[TPW] = {};
    // prologue: fragments for k-step 0
    v8h a0 = *(const v8h*)(a);
    v8h a1 = *(const v8h*)(a + 16);
    v8h b0[TPW], b1[TPW];
    #pragma unroll
    for (int j = 0; j < TPW; ++j) {
        const _Float16* wp = wbase + (size_t)j * 16 * KC;
        b0[j] = *(const v8h*)(wp);
        b1[j] = *(const v8h*)(wp + 16);
    }
    // pipelined body: load step ks+1, multiply step ks
    #pragma unroll
    for (int ks = 0; ks < NK - 1; ++ks) {
        const int k0 = (ks + 1) * 32;
        v8h na0 = *(const v8h*)(a + k0);
        v8h na1 = *(const v8h*)(a + k0 + 16);
        v8h nb0[TPW], nb1[TPW];
        #pragma unroll
        for (int j = 0; j < TPW; ++j) {
            const _Float16* wp = wbase + (size_t)j * 16 * KC + k0;
            nb0[j] = *(const v8h*)(wp);
            nb1[j] = *(const v8h*)(wp + 16);
        }
        v16h af = cat16(a0, a1);
        #pragma unroll
        for (int j = 0; j < TPW; ++j) {
            acc[j] = __builtin_amdgcn_wmma_f32_16x16x32_f16(false, af, false, cat16(b0[j], b1[j]),
                                                            (short)0, acc[j], false, false);
        }
        a0 = na0; a1 = na1;
        #pragma unroll
        for (int j = 0; j < TPW; ++j) { b0[j] = nb0[j]; b1[j] = nb1[j]; }
    }
    // epilogue: last k-step
    {
        v16h af = cat16(a0, a1);
        #pragma unroll
        for (int j = 0; j < TPW; ++j) {
            acc[j] = __builtin_amdgcn_wmma_f32_16x16x32_f16(false, af, false, cat16(b0[j], b1[j]),
                                                            (short)0, acc[j], false, false);
        }
    }
    // C layout: VGPR i -> row tm*16 + hs*8 + i, col = panel + j*16 + r
    const size_t row0 = (size_t)(tm * 16 + hs * 8);
    const int col0 = blockIdx.y * ROWS + r;
    #pragma unroll
    for (int j = 0; j < TPW; ++j) {
        #pragma unroll
        for (int i = 0; i < 8; ++i) {
            float v = acc[j][i];
            if (POSTOP) v = (v != v) ? 0.0f : clip1e6(v);
            C[(row0 + i) * N + col0 + j * 16] = (OUT_T)v;
        }
    }
}

// ---------------- setup kernels ----------------
__global__ void f32_to_f16_kernel(const float* __restrict__ s, _Float16* __restrict__ d, int n) {
    int i = blockIdx.x * blockDim.x + threadIdx.x;
    if (i < n) d[i] = (_Float16)s[i];
}

// (12,44,384) -> (12,48,384), zero-padded rows 44..47
__global__ void pack_xpw_kernel(const float* __restrict__ s, _Float16* __restrict__ d) {
    int i = blockIdx.x * blockDim.x + threadIdx.x;
    if (i >= DEPTH * XP_N * D_INNER) return;
    int k = i % D_INNER;
    int n = (i / D_INNER) % XP_N;
    int l = i / (D_INNER * XP_N);
    d[i] = (n < 44) ? (_Float16)s[((size_t)l * 44 + n) * D_INNER + k] : (_Float16)0.0f;
}

// im2col of (8,3,8,224,224) video into (12544, 768) f16; k order = (c,i,j)
__global__ void im2col_kernel(const float* __restrict__ x, _Float16* __restrict__ A16) {
    int idx = blockIdx.x * blockDim.x + threadIdx.x;
    if (idx >= M_PATCH * K_PATCH) return;
    int m = idx / K_PATCH, k = idx % K_PATCH;
    int b = m / (T_FR * NP_);
    int r = m % (T_FR * NP_);
    int t = r / NP_, p = r % NP_;
    int py = p / 14, px = p % 14;
    int c = k / 256, q = k % 256, ii = q / 16, jj = q % 16;
    size_t src = ((((size_t)b * 3 + c) * T_FR + t) * 224 + (py * 16 + ii)) * 224 + (px * 16 + jj);
    A16[idx] = (_Float16)x[src];
}

// tokens = [cls | patches] + pos_embed + temp_pos ; res = 0
__global__ void assemble_kernel(const float* __restrict__ pe, const float* __restrict__ patch_b,
                                const float* __restrict__ cls, const float* __restrict__ pos,
                                const float* __restrict__ tpos,
                                float* __restrict__ h, float* __restrict__ res) {
    int idx = blockIdx.x * blockDim.x + threadIdx.x;
    if (idx >= M_TOK * D_MODEL) return;
    int m = idx / D_MODEL, d = idx % D_MODEL;
    int b = m / L_SEQ, s = m % L_SEQ;
    int t = s / (NP_ + 1), pi = s % (NP_ + 1);
    float v = pos[pi * D_MODEL + d] + tpos[t * D_MODEL + d];
    if (pi == 0) {
        v += cls[t * D_MODEL + d];
    } else {
        v += pe[(((size_t)b * T_FR + t) * NP_ + (pi - 1)) * D_MODEL + d] + patch_b[d];
    }
    h[idx] = v;
    res[idx] = 0.0f;
}

// ---------------- per-layer kernels ----------------
// res = clip(clip(h)+clip(res)); hn16 = f16(LN(res))   -- one wave32 per token
__global__ void ln_res_kernel(const float* __restrict__ h, float* __restrict__ res,
                              _Float16* __restrict__ hn16,
                              const float* __restrict__ nw, const float* __restrict__ nb) {
    int wid  = (blockIdx.x * blockDim.x + threadIdx.x) >> 5;
    int lane = threadIdx.x & 31;
    if (wid >= M_TOK) return;
    size_t base = (size_t)wid * D_MODEL;
    float v[6];
    float s = 0.0f;
    #pragma unroll
    for (int e = 0; e < 6; ++e) {
        int k = e * 32 + lane;
        float t = clip1e6(clip1e6(h[base + k]) + clip1e6(res[base + k]));
        res[base + k] = t;
        v[e] = t;
        s += t;
    }
    float mean = wave_red(s) * (1.0f / D_MODEL);
    float q = 0.0f;
    #pragma unroll
    for (int e = 0; e < 6; ++e) { float d = v[e] - mean; q += d * d; }
    float rstd = rsqrtf(wave_red(q) * (1.0f / D_MODEL) + 1e-5f);
    #pragma unroll
    for (int e = 0; e < 6; ++e) {
        int k = e * 32 + lane;
        hn16[base + k] = (_Float16)((v[e] - mean) * rstd * nw[k] + nb[k]);
    }
}

// depthwise causal conv(k=4) + SiLU over branch-ordered sequence.
// xp lives in xz16 cols [0,384); `reverse` maps branch index -> original index.
__global__ void conv_silu_kernel(const _Float16* __restrict__ xz16,
                                 const float* __restrict__ cw, const float* __restrict__ cb,
                                 float* __restrict__ xc, _Float16* __restrict__ xc16,
                                 int reverse) {
    int idx = blockIdx.x * blockDim.x + threadIdx.x;
    if (idx >= M_TOK * D_INNER) return;
    int d = idx % D_INNER;
    int m = idx / D_INNER;
    int b = m / L_SEQ, l = m % L_SEQ;
    float acc = cb[d];
    #pragma unroll
    for (int j = 0; j < 4; ++j) {
        int i = l - 3 + j;
        if (i >= 0) {
            int lo = reverse ? (L_SEQ - 1 - i) : i;
            acc += cw[d * 4 + j] * (float)xz16[((size_t)(b * L_SEQ + lo)) * (2 * D_INNER) + d];
        }
    }
    float sv = acc / (1.0f + __expf(-acc));
    xc[idx]   = sv;
    xc16[idx] = (_Float16)sv;
}

// dt = softplus(xdbl[:, :12] @ dtw.T + dtb)
__global__ void dt_kernel(const float* __restrict__ xdbl, const float* __restrict__ dtw,
                          const float* __restrict__ dtb, float* __restrict__ out) {
    int idx = blockIdx.x * blockDim.x + threadIdx.x;
    if (idx >= M_TOK * D_INNER) return;
    int d = idx % D_INNER;
    size_t m = idx / D_INNER;
    float a = dtb[d];
    const float* row = xdbl + m * XP_N;
    #pragma unroll
    for (int j = 0; j < DT_RANK; ++j) a += row[j] * dtw[d * DT_RANK + j];
    out[idx] = (a > 20.0f) ? a : log1pf(__expf(a));
}

// selective scan + D-skip + SiLU(z) gate.  One thread per (b,d); 16 states in regs.
__global__ void scan_kernel(const float* __restrict__ xc, const float* __restrict__ dtbuf,
                            const float* __restrict__ xdbl, const float* __restrict__ alog,
                            const float* __restrict__ Dp, const _Float16* __restrict__ xz16,
                            float* __restrict__ ycomb, int reverse, int accumulate) {
    int gid = blockIdx.x * blockDim.x + threadIdx.x;
    if (gid >= BATCH * D_INNER) return;
    int b = gid / D_INNER, d = gid % D_INNER;
    float A[N_STATE], hreg[N_STATE];
    #pragma unroll
    for (int n = 0; n < N_STATE; ++n) {
        A[n] = -__expf(alog[d * N_STATE + n]);
        hreg[n] = 0.0f;
    }
    float dpd = Dp[d];
    for (int t = 0; t < L_SEQ; ++t) {
        size_t mb = (size_t)b * L_SEQ + t;            // branch order
        float dt = dtbuf[mb * D_INNER + d];
        float u  = xc[mb * D_INNER + d];
        const float* bc = xdbl + mb * XP_N;
        __builtin_prefetch(bc + XP_N, 0, 1);          // next token's B/C row
        float du = dt * u, y = 0.0f;
        #pragma unroll
        for (int n = 0; n < N_STATE; ++n) {
            float da = __expf(dt * A[n]);
            hreg[n] = da * hreg[n] + du * bc[DT_RANK + n];
            y += hreg[n] * bc[DT_RANK + N_STATE + n];
        }
        y += u * dpd;
        int lo = reverse ? (L_SEQ - 1 - t) : t;       // original order
        size_t mo = (size_t)b * L_SEQ + lo;
        float z = (float)xz16[mo * (2 * D_INNER) + D_INNER + d];
        y *= z / (1.0f + __expf(-z));
        if (accumulate) ycomb[mo * D_INNER + d] += y;
        else            ycomb[mo * D_INNER + d]  = y;
    }
}

// final: LN(clip(h+res)) on cls tokens, then head GEMV (one block per (b,t))
__global__ void head_kernel(const float* __restrict__ h, const float* __restrict__ res,
                            const float* __restrict__ nfw, const float* __restrict__ nfb,
                            const float* __restrict__ hw, const float* __restrict__ hb,
                            float* __restrict__ out) {
    __shared__ float sv[D_MODEL];
    __shared__ float stats[2];
    int bt = blockIdx.x;
    int b = bt / T_FR, t = bt % T_FR;
    size_t m = ((size_t)b * L_SEQ + t * (NP_ + 1)) * D_MODEL;
    int k = threadIdx.x;
    float v = clip1e6(h[m + k] + res[m + k]);
    sv[k] = v;
    __syncthreads();
    if (k == 0) {
        float s = 0.0f;
        for (int i = 0; i < D_MODEL; ++i) s += sv[i];
        float mean = s / D_MODEL;
        float q = 0.0f;
        for (int i = 0; i < D_MODEL; ++i) { float d = sv[i] - mean; q += d * d; }
        stats[0] = mean;
        stats[1] = rsqrtf(q / D_MODEL + 1e-5f);
    }
    __syncthreads();
    float hf = (v - stats[0]) * stats[1] * nfw[k] + nfb[k];
    sv[k] = hf;
    __syncthreads();
    if (k < NC_) {
        float a = hb[k];
        for (int i = 0; i < D_MODEL; ++i) a += sv[i] * hw[k * D_MODEL + i];
        out[(b * T_FR + t) * NC_ + k] = a;
    }
}

// ---------------- host side ----------------
static constexpr size_t algn(size_t x) { return (x + 255) & ~(size_t)255; }

extern "C" void kernel_launch(void* const* d_in, const int* in_sizes, int n_in,
                              void* d_out, int out_size, void* d_ws, size_t ws_size,
                              hipStream_t stream) {
    const float* x        = (const float*)d_in[0];
    const float* patch_w  = (const float*)d_in[1];
    const float* patch_b  = (const float*)d_in[2];
    const float* cls_tok  = (const float*)d_in[3];
    const float* pos_emb  = (const float*)d_in[4];
    const float* temp_pos = (const float*)d_in[5];
    const float* norm_w   = (const float*)d_in[6];
    const float* norm_b   = (const float*)d_in[7];
    const float* in_projw = (const float*)d_in[8];
    const float* conv_w   = (const float*)d_in[9];
    const float* conv_b   = (const float*)d_in[10];
    const float* x_projw  = (const float*)d_in[11];
    const float* dt_w     = (const float*)d_in[12];
    const float* dt_b     = (const float*)d_in[13];
    const float* A_log    = (const float*)d_in[14];
    const float* Dp       = (const float*)d_in[15];
    const float* conv_wr  = (const float*)d_in[16];
    const float* conv_br  = (const float*)d_in[17];
    const float* x_projwr = (const float*)d_in[18];
    const float* dt_wr    = (const float*)d_in[19];
    const float* dt_br    = (const float*)d_in[20];
    const float* A_logr   = (const float*)d_in[21];
    const float* Dpr      = (const float*)d_in[22];
    const float* out_projw= (const float*)d_in[23];
    const float* normf_w  = (const float*)d_in[24];
    const float* normf_b  = (const float*)d_in[25];
    const float* head_w   = (const float*)d_in[26];
    const float* head_b   = (const float*)d_in[27];
    float* out = (float*)d_out;

    // workspace layout
    char* ws = (char*)d_ws;
    size_t off = 0;
    auto grab = [&](size_t bytes) { char* p = ws + off; off += algn(bytes); return p; };
    _Float16* A16    = (_Float16*)grab((size_t)M_PATCH * K_PATCH * 2);
    _Float16* Wpat16 = (_Float16*)grab((size_t)D_MODEL * K_PATCH * 2);
    _Float16* Wip16  = (_Float16*)grab((size_t)DEPTH * 2 * D_INNER * D_MODEL * 2);
    _Float16* Wop16  = (_Float16*)grab((size_t)DEPTH * D_MODEL * D_INNER * 2);
    _Float16* Wxp16  = (_Float16*)grab((size_t)DEPTH * XP_N * D_INNER * 2);
    _Float16* Wxpr16 = (_Float16*)grab((size_t)DEPTH * XP_N * D_INNER * 2);
    float*    pe     = (float*)   grab((size_t)M_PATCH * D_MODEL * 4);
    float*    hbuf   = (float*)   grab((size_t)M_TOK * D_MODEL * 4);
    float*    resbuf = (float*)   grab((size_t)M_TOK * D_MODEL * 4);
    _Float16* hn16   = (_Float16*)grab((size_t)M_TOK * D_MODEL * 2);
    _Float16* xz16   = (_Float16*)grab((size_t)M_TOK * 2 * D_INNER * 2);
    float*    xcbuf  = (float*)   grab((size_t)M_TOK * D_INNER * 4);
    _Float16* xc16   = (_Float16*)grab((size_t)M_TOK * D_INNER * 2);
    float*    xdbl   = (float*)   grab((size_t)M_TOK * XP_N * 4);
    float*    dtbuf  = (float*)   grab((size_t)M_TOK * D_INNER * 4);
    float*    ycomb  = (float*)   grab((size_t)M_TOK * D_INNER * 4);
    _Float16* yc16   = (_Float16*)grab((size_t)M_TOK * D_INNER * 2);
    (void)ws_size; (void)in_sizes; (void)n_in; (void)out_size;

    auto cdiv = [](long long a, long long b) { return (int)((a + b - 1) / b); };

    // ---- setup: weight conversion, im2col, patch embed, token assembly ----
    {
        int n;
        n = D_MODEL * K_PATCH;
        f32_to_f16_kernel<<<cdiv(n, 256), 256, 0, stream>>>(patch_w, Wpat16, n);
        n = DEPTH * 2 * D_INNER * D_MODEL;
        f32_to_f16_kernel<<<cdiv(n, 256), 256, 0, stream>>>(in_projw, Wip16, n);
        n = DEPTH * D_MODEL * D_INNER;
        f32_to_f16_kernel<<<cdiv(n, 256), 256, 0, stream>>>(out_projw, Wop16, n);
        n = DEPTH * XP_N * D_INNER;
        pack_xpw_kernel<<<cdiv(n, 256), 256, 0, stream>>>(x_projw, Wxp16);
        pack_xpw_kernel<<<cdiv(n, 256), 256, 0, stream>>>(x_projwr, Wxpr16);
        n = M_PATCH * K_PATCH;
        im2col_kernel<<<cdiv(n, 256), 256, 0, stream>>>(x, A16);
        // patch embed: M=12544, N=192, K=768 -> TPW=2 (panel 32x768 = 48KB)
        {
            dim3 g(cdiv(M_PATCH / 16, 8), D_MODEL / 32);
            gemm_wmma_kernel<float, 0, 2, K_PATCH><<<g, 256, 0, stream>>>(
                A16, Wpat16, pe, M_PATCH, D_MODEL);
        }
        n = M_TOK * D_MODEL;
        assemble_kernel<<<cdiv(n, 256), 256, 0, stream>>>(pe, patch_b, cls_tok, pos_emb,
                                                          temp_pos, hbuf, resbuf);
    }

    const int mgrid = cdiv(M_TOK / 16, 8);   // 99

    // ---- 12 layers ----
    for (int l = 0; l < DEPTH; ++l) {
        ln_res_kernel<<<cdiv((long long)M_TOK * 32, 256), 256, 0, stream>>>(
            hbuf, resbuf, hn16, norm_w + l * D_MODEL, norm_b + l * D_MODEL);

        // in_proj: N=768, K=192 -> TPW=4 (panel 64x192 = 24KB)
        {
            dim3 g(mgrid, (2 * D_INNER) / 64);
            gemm_wmma_kernel<_Float16, 0, 4, D_MODEL><<<g, 256, 0, stream>>>(
                hn16, Wip16 + (size_t)l * 2 * D_INNER * D_MODEL, xz16,
                M_TOK, 2 * D_INNER);
        }

        const int nElems = M_TOK * D_INNER;
        for (int dir = 0; dir < 2; ++dir) {
            const float* cw  = dir ? conv_wr  + (size_t)l * D_INNER * 4       : conv_w  + (size_t)l * D_INNER * 4;
            const float* cb  = dir ? conv_br  + (size_t)l * D_INNER           : conv_b  + (size_t)l * D_INNER;
            const _Float16* wxp = dir ? Wxpr16 + (size_t)l * XP_N * D_INNER   : Wxp16   + (size_t)l * XP_N * D_INNER;
            const float* dw  = dir ? dt_wr    + (size_t)l * D_INNER * DT_RANK : dt_w    + (size_t)l * D_INNER * DT_RANK;
            const float* db  = dir ? dt_br    + (size_t)l * D_INNER           : dt_b    + (size_t)l * D_INNER;
            const float* al  = dir ? A_logr   + (size_t)l * D_INNER * N_STATE : A_log   + (size_t)l * D_INNER * N_STATE;
            const float* dpp = dir ? Dpr      + (size_t)l * D_INNER           : Dp      + (size_t)l * D_INNER;

            conv_silu_kernel<<<cdiv(nElems, 256), 256, 0, stream>>>(
                xz16, cw, cb, xcbuf, xc16, dir);
            // x_proj: N=48, K=384 -> TPW=3 (panel 48x384 = 36KB), one N panel
            {
                dim3 g(mgrid, 1);
                gemm_wmma_kernel<float, 0, 3, D_INNER><<<g, 256, 0, stream>>>(
                    xc16, wxp, xdbl, M_TOK, XP_N);
            }
            dt_kernel<<<cdiv(nElems, 256), 256, 0, stream>>>(xdbl, dw, db, dtbuf);
            scan_kernel<<<cdiv(BATCH * D_INNER, 128), 128, 0, stream>>>(
                xcbuf, dtbuf, xdbl, al, dpp, xz16, ycomb, dir, dir);
        }

        f32_to_f16_kernel<<<cdiv(nElems, 256), 256, 0, stream>>>(ycomb, yc16, nElems);
        // out_proj: N=192, K=384 -> TPW=4 (panel 64x384 = 48KB)
        {
            dim3 g(mgrid, D_MODEL / 64);
            gemm_wmma_kernel<float, 1, 4, D_INNER><<<g, 256, 0, stream>>>(
                yc16, Wop16 + (size_t)l * D_MODEL * D_INNER, hbuf,
                M_TOK, D_MODEL);
        }
    }

    // ---- final LN + head on cls tokens ----
    head_kernel<<<BATCH * T_FR, D_MODEL, 0, stream>>>(hbuf, resbuf, normf_w, normf_b,
                                                      head_w, head_b, out);
}
